// GNN_85787676770672
// MI455X (gfx1250) — matrix operand order repaired and checked by reading, not compiled, not run on hardware
//
#include <hip/hip_runtime.h>
#include <hip/hip_bf16.h>
#include <math.h>

typedef __attribute__((ext_vector_type(16))) _Float16 v16h;
typedef __attribute__((ext_vector_type(8)))  float    v8f;

#define BATCH  4
#define CCH    128
#define GH     128
#define GW     128
#define NN     (GH*GW)          // 16384 = 2^14
#define HEADS  8
#define DHEAD  16

__device__ __forceinline__ unsigned short f32_to_f16u(float f) {
  union { _Float16 h; unsigned short u; } cv;
  cv.h = (_Float16)f;
  return cv.u;
}

// LDS aperture addresses keep the LDS byte offset in addr[31:0] (ISA 10.2),
// so truncating a generic pointer to a shared
// variable yields the LDS offset the async-DMA instructions expect.
__device__ __forceinline__ unsigned lds_off(const void* p) {
  return (unsigned)(unsigned long long)p;
}

// Per-lane 16B async copy global -> LDS, tracked by ASYNCcnt.
__device__ __forceinline__ void async_copy_b128(unsigned lds_addr,
                                                unsigned voff,
                                                unsigned long long gbase) {
  asm volatile("global_load_async_to_lds_b128 %0, %1, %2"
               :
               : "v"(lds_addr), "v"(voff), "s"(gbase)
               : "memory");
}

__device__ __forceinline__ void wait_asynccnt0() {
  asm volatile("s_wait_asynccnt 0x0" ::: "memory");
}

// ---------------------------------------------------------------------------
// Kernel 0 (prep, runs once, 64 blocks): W[c][out] f32 -> Wt16[out][c] f16.
// Amortizes the convert+transpose so every GEMM block can async-DMA Wt raw.
// ---------------------------------------------------------------------------
__global__ __launch_bounds__(256)
void gat_prep_w(const float* __restrict__ Wlin,
                unsigned short* __restrict__ Wt16)
{
  int e = blockIdx.x*256 + threadIdx.x;      // 16384 elements
  int c = e >> 7, o = e & 127;
  Wt16[o*CCH + c] = f32_to_f16u(Wlin[e]);
}

// ---------------------------------------------------------------------------
// Kernel 1: xp[b,n,:] = x[b,:,n]^T @ W via WMMA f16->f32; fused per-node
// attention logits. Staging of both LDS tiles uses async global->LDS DMA.
// Block = 256 threads (8 waves); wave w computes the 16x16 tile of head w.
// ---------------------------------------------------------------------------
__global__ __launch_bounds__(256)
void gat_linear_wmma(const float* __restrict__ x,
                     const unsigned short* __restrict__ Wt16,
                     const float* __restrict__ att_src,
                     const float* __restrict__ att_dst,
                     float* __restrict__ xp,
                     float* __restrict__ asrc,
                     float* __restrict__ adst)
{
  __shared__ __align__(16) unsigned short Wt[CCH*CCH]; // [out][c] f16 (32 KB)
  __shared__ __align__(16) float          Af[CCH*16];  // [c][m]   f32 ( 8 KB)

  const int t  = threadIdx.x;
  const int b  = blockIdx.y;
  const int n0 = blockIdx.x * 16;

  // ---- async stage Wt (32768 B = 2048 x 16 B segments, 8 per thread) ----
  {
    const unsigned wbase = lds_off(Wt);
    const unsigned long long g = (unsigned long long)Wt16;
    #pragma unroll
    for (int i = 0; i < 8; ++i) {
      unsigned seg = (unsigned)t + i*256u;
      async_copy_b128(wbase + seg*16u, seg*16u, g);
    }
  }
  // ---- async stage x-slab: x[b, c, n0+m], 2048 floats = 512 segments ----
  {
    const unsigned abase = lds_off(Af);
    const unsigned long long g = (unsigned long long)x;
    #pragma unroll
    for (int i = 0; i < 2; ++i) {
      unsigned seg = (unsigned)t + i*256u;
      unsigned c = seg >> 2, q = seg & 3;            // 4 x 16B per c-row
      unsigned voff = (((unsigned)(b*CCH) + c)*(unsigned)NN + (unsigned)n0 + q*4u)*4u;
      async_copy_b128(abase + (c*16u + q*4u)*4u, voff, g);
    }
  }
  wait_asynccnt0();
  __syncthreads();

  const int lane = t & 31;
  const int wave = t >> 5;        // head index / 16-col tile
  const int half = lane >> 4;     // K-half selector (wave32 layout)
  const int ncol = lane & 15;     // B/C column
  const int mrow = lane & 15;     // A row

  const unsigned* W32 = (const unsigned*)Wt;

  union Frag { v16h h; unsigned u[8]; };
  Frag fa, fb;
  v8f acc = {};

  #pragma unroll
  for (int k0 = 0; k0 < CCH; k0 += 32) {
    #pragma unroll
    for (int r = 0; r < 8; ++r) {
      // A 16x32 f16: lanes 0-15 hold K {0..7,16..23}, lanes 16-31 +8.
      int klocal = (r < 4) ? (2*r) : (16 + 2*(r-4));
      int ka = k0 + klocal + 8*half;
      union { _Float16 h[2]; unsigned u; } pk;
      pk.h[0] = (_Float16)Af[(ka+0)*16 + mrow];
      pk.h[1] = (_Float16)Af[(ka+1)*16 + mrow];
      fa.u[r] = pk.u;
      // B 32x16 f16: lane = N col; lanes 0-15 hold K {0..7,16..23}, lanes 16-31 +8.
      int kb = k0 + klocal + 8*half;
      fb.u[r] = W32[(wave*16 + ncol)*(CCH/2) + (kb >> 1)];
    }
    acc = __builtin_amdgcn_wmma_f32_16x16x32_f16(false, fa.h, false, fb.h,
                                                 (short)0, acc, false, false);
  }

  // Epilogue: store xp tile, fuse a_src/a_dst = <xp[n,h,:], att_{src,dst}[h]>.
  const float as_w = att_src[wave*DHEAD + ncol];
  const float ad_w = att_dst[wave*DHEAD + ncol];
  #pragma unroll
  for (int r = 0; r < 8; ++r) {
    int row = r + 8*half;                    // C/D layout: VGPR r -> M=r(+8)
    float v = acc[r];
    xp[((size_t)(b*NN) + n0 + row)*CCH + wave*16 + ncol] = v;
    float ps = v * as_w, pd = v * ad_w;
    #pragma unroll
    for (int m = 8; m >= 1; m >>= 1) {       // reduce over 16 N-lanes
      ps += __shfl_xor(ps, m, 16);
      pd += __shfl_xor(pd, m, 16);
    }
    if (ncol == 0) {
      asrc[((size_t)(b*NN) + n0 + row)*HEADS + wave] = ps;
      adst[((size_t)(b*NN) + n0 + row)*HEADS + wave] = pd;
    }
  }
}

// ---------------------------------------------------------------------------
// Kernel 2 (fused): attention softmax+aggregation over the static 3x3
// neighborhood (+ duplicated self loop), then bias+ELU+LayerNorm+transpose,
// all in one pass. Block = 256 threads = 32 nodes x 8 heads; each thread's
// 16 aggregated channels are exactly its LN slice, so mean/var reduces over
// registers with width-8 shuffles; LDS only does the channel<->node transpose.
// Saves the 67 MB agg global round-trip of the unfused version.
// ---------------------------------------------------------------------------
__global__ __launch_bounds__(256)
void gat_attn_ln(const float* __restrict__ xp,
                 const float* __restrict__ asrc,
                 const float* __restrict__ adst,
                 const float* __restrict__ bias,
                 const float* __restrict__ gamma,
                 const float* __restrict__ beta,
                 float* __restrict__ out)
{
  __shared__ float tile[32][CCH + 1];
  __shared__ float mu_s[32], rs_s[32];

  const int t  = threadIdx.x;
  const int b  = blockIdx.y;
  const int n0 = blockIdx.x * 32;
  const int j  = t >> 3;                 // local node
  const int h  = t & 7;                  // head (fastest -> coalesced gathers)
  const int n  = n0 + j;
  const int r  = n >> 7, cc = n & 127;

  // ---- attention scores over <=9 grid neighbors + duplicated self loop ----
  const float ad = adst[((size_t)(b*NN) + n)*HEADS + h];
  float sc[10]; int srcs[10]; int cnt = 0;
  float mmax = -1e30f;
  #pragma unroll
  for (int dr = -1; dr <= 1; ++dr) {
    #pragma unroll
    for (int dc = -1; dc <= 1; ++dc) {
      int nr = r + dr, nc = cc + dc;
      if ((unsigned)nr < GH && (unsigned)nc < GW) {
        int s = nr*GW + nc;
        float v = asrc[((size_t)(b*NN) + s)*HEADS + h] + ad;
        v = v > 0.f ? v : 0.2f*v;        // leaky_relu(0.2)
        srcs[cnt] = s; sc[cnt] = v;
        mmax = fmaxf(mmax, v);
        ++cnt;
      }
    }
  }
  { // add_self_loops appends a second self edge (same score -> weight 2e)
    float v = asrc[((size_t)(b*NN) + n)*HEADS + h] + ad;
    v = v > 0.f ? v : 0.2f*v;
    srcs[cnt] = n; sc[cnt] = v; ++cnt;
  }

  float den = 0.f;
  for (int i = 0; i < cnt; ++i) { sc[i] = __expf(sc[i] - mmax); den += sc[i]; }
  const float inv = 1.f / den;

  // ---- alpha-weighted aggregation of xp[src, h, :] ----
  float accv[16];
  #pragma unroll
  for (int d = 0; d < 16; ++d) accv[d] = 0.f;
  for (int i = 0; i < cnt; ++i) {
    float w = sc[i] * inv;
    const float4* p =
        (const float4*)(xp + ((size_t)(b*NN) + srcs[i])*CCH + h*DHEAD);
    #pragma unroll
    for (int q = 0; q < 4; ++q) {
      float4 v = p[q];
      accv[4*q+0] = fmaf(w, v.x, accv[4*q+0]);
      accv[4*q+1] = fmaf(w, v.y, accv[4*q+1]);
      accv[4*q+2] = fmaf(w, v.z, accv[4*q+2]);
      accv[4*q+3] = fmaf(w, v.w, accv[4*q+3]);
    }
  }

  // ---- bias + ELU + per-node stats (registers), stash in LDS transpose ----
  float sum = 0.f, sq = 0.f;
  #pragma unroll
  for (int d = 0; d < 16; ++d) {
    float v = accv[d] + bias[h*DHEAD + d];
    v = v > 0.f ? v : (__expf(v) - 1.f);   // ELU
    tile[j][h*DHEAD + d] = v;
    sum += v; sq += v*v;
  }
  #pragma unroll
  for (int m = 4; m >= 1; m >>= 1) {       // 8 threads of node j are adjacent
    sum += __shfl_xor(sum, m, 8);
    sq  += __shfl_xor(sq,  m, 8);
  }
  if (h == 0) {
    float mu  = sum * (1.f/CCH);
    float var = sq  * (1.f/CCH) - mu*mu;   // population var (ddof=0)
    mu_s[j] = mu;
    rs_s[j] = rsqrtf(var + 1e-5f);
  }
  __syncthreads();

  // ---- normalize + transposed coalesced store (lane -> consecutive n) ----
  {
    int jj = t & 31, cb = t >> 5;
    float mu = mu_s[jj], rs = rs_s[jj];
    for (int c = cb; c < CCH; c += 8) {
      float v = (tile[jj][c] - mu) * rs * gamma[c] + beta[c];
      out[(((size_t)(b*CCH) + c) << 14) + n0 + jj] = v;
    }
  }
}

// ---------------------------------------------------------------------------
extern "C" void kernel_launch(void* const* d_in, const int* in_sizes, int n_in,
                              void* d_out, int out_size, void* d_ws, size_t ws_size,
                              hipStream_t stream)
{
  (void)in_sizes; (void)n_in; (void)out_size; (void)ws_size;

  const float* x       = (const float*)d_in[0];   // [B,C,H,W]
  const float* Wlin    = (const float*)d_in[1];   // [C, H*D]
  const float* att_src = (const float*)d_in[2];   // [HEADS, DHEAD]
  const float* att_dst = (const float*)d_in[3];
  const float* bias    = (const float*)d_in[4];   // [C]
  const float* gamma   = (const float*)d_in[5];
  const float* beta    = (const float*)d_in[6];
  float* out = (float*)d_out;                     // [B,C,H,W]

  char* ws = (char*)d_ws;
  const size_t WT_BYTES = (size_t)CCH*CCH*sizeof(unsigned short); // 32 KB
  const size_t XP_BYTES = (size_t)BATCH*NN*CCH*sizeof(float);     // 33.5 MB
  const size_t A_BYTES  = (size_t)BATCH*NN*HEADS*sizeof(float);   //  2.1 MB
  unsigned short* Wt16 = (unsigned short*)(ws);
  float* xp   = (float*)(ws + WT_BYTES);
  float* asrc = (float*)(ws + WT_BYTES + XP_BYTES);
  float* adst = (float*)(ws + WT_BYTES + XP_BYTES + A_BYTES);

  gat_prep_w<<<dim3((CCH*CCH)/256), 256, 0, stream>>>(Wlin, Wt16);

  gat_linear_wmma<<<dim3(NN/16, BATCH), 256, 0, stream>>>(
      x, Wt16, att_src, att_dst, xp, asrc, adst);

  gat_attn_ln<<<dim3(NN/32, BATCH), 256, 0, stream>>>(
      xp, asrc, adst, bias, gamma, beta, out);
}
